// Attention_37125697306831
// MI455X (gfx1250) — matrix operand order repaired and checked by reading, not compiled, not run on hardware
//
#include <hip/hip_runtime.h>
#include <hip/hip_bf16.h>

typedef __bf16 bf16;
typedef __attribute__((ext_vector_type(16))) __bf16 v16bf;
typedef __attribute__((ext_vector_type(8)))  float  v8f;

#define C_DIM   256
#define NHEADS  8
#define HDIM    32
#define NSEQ    2304
#define BATCH   4
#define MTOT    (BATCH * NSEQ)   // 9216 rows total

// ---------------------------------------------------------------------------
// WMMA fragment helpers (CDNA5 wave32 layouts, CDNA5 ISA §7.12.2)
// ---------------------------------------------------------------------------

// A 16x32 bf16: lane L holds row M=L&15; lane-half selects K groups.
__device__ __forceinline__ int a_k_of(int j, int hi) {
  return 16 * hi + j;
}

__device__ __forceinline__ v16bf load_a_f32(const float* base, int ld, int lane) {
  int m  = lane & 15;
  int hi = lane >> 4;
  const float* row = base + m * ld;
  v16bf a;
#pragma unroll
  for (int j = 0; j < 16; ++j) a[j] = (bf16)row[a_k_of(j, hi)];
  return a;
}

__device__ __forceinline__ v16bf load_a_bf16(const bf16* base, int ld, int lane) {
  int m  = lane & 15;
  int hi = lane >> 4;
  const bf16* row = base + m * ld;
  v16bf a;
#pragma unroll
  for (int j = 0; j < 16; ++j) a[j] = row[a_k_of(j, hi)];
  return a;
}

// B 32x16 (KxN): lane L holds column N=L&15; K range = (L>>4)*16 .. +15.
// element (k,n) read from base[n*ld + k]  (k contiguous in memory)
__device__ __forceinline__ v16bf load_b_rowmajor_f32(const float* base, int ld, int lane) {
  int n    = lane & 15;
  int koff = (lane >> 4) * 16;
  const float* p = base + n * ld + koff;
  v16bf b;
#pragma unroll
  for (int j = 0; j < 16; ++j) b[j] = (bf16)p[j];
  return b;
}

__device__ __forceinline__ v16bf load_b_rowmajor_bf16(const bf16* base, int ld, int lane) {
  int n    = lane & 15;
  int koff = (lane >> 4) * 16;
  const bf16* p = base + n * ld + koff;
  v16bf b;
#pragma unroll
  for (int j = 0; j < 16; ++j) b[j] = p[j];
  return b;
}

// element (k,n) read from base[k*ld + n]  (k strided by ld)
__device__ __forceinline__ v16bf load_b_colmajor_bf16(const bf16* base, int ld, int lane) {
  int n    = lane & 15;
  int koff = (lane >> 4) * 16;
  v16bf b;
#pragma unroll
  for (int j = 0; j < 16; ++j) b[j] = base[(koff + j) * ld + n];
  return b;
}

__device__ __forceinline__ v8f wmma_bf16(v16bf a, v16bf b, v8f c) {
  return __builtin_amdgcn_wmma_f32_16x16x32_bf16(
      false, a, false, b, (short)0, c, false, false);
}

// ---------------------------------------------------------------------------
// Kernel 1: projection GEMM  out[m,n] = sum_k A[m,k] * W[n,k] (+ bias[n])
// Each wave computes a 16x64 strip (4 accumulators, A-fragment reused 4x).
// ---------------------------------------------------------------------------
__global__ __launch_bounds__(256)
void proj_f32_to_bf16(const float* __restrict__ A, const float* __restrict__ W,
                      const float* __restrict__ bias, bf16* __restrict__ out) {
  int lane = threadIdx.x & 31;
  int wave = threadIdx.x >> 5;
  int gw   = blockIdx.x * 8 + wave;    // 2304 strips total
  int tn   = gw & 3;                   // 4 strips of 64 cols
  int tm   = gw >> 2;                  // 576 row tiles

  v8f acc[4] = {{}, {}, {}, {}};
#pragma unroll
  for (int k0 = 0; k0 < C_DIM; k0 += 32) {
    v16bf af = load_a_f32(A + (size_t)tm * 16 * C_DIM + k0, C_DIM, lane);
#pragma unroll
    for (int s = 0; s < 4; ++s) {
      v16bf bf = load_b_rowmajor_f32(W + (size_t)(tn * 64 + s * 16) * C_DIM + k0,
                                     C_DIM, lane);
      acc[s] = wmma_bf16(af, bf, acc[s]);
    }
  }

  int n  = lane & 15;
  int hi = lane >> 4;
#pragma unroll
  for (int s = 0; s < 4; ++s) {
    int col  = tn * 64 + s * 16 + n;
    float bv = bias ? bias[col] : 0.0f;
#pragma unroll
    for (int r = 0; r < 8; ++r) {
      int m = tm * 16 + r + 8 * hi;
      out[(size_t)m * C_DIM + col] = (bf16)(acc[s][r] + bv);
    }
  }
}

// ---------------------------------------------------------------------------
// Kernel 2: final projection, A bf16 (attn output), out fp32 (d_out)
// ---------------------------------------------------------------------------
__global__ __launch_bounds__(256)
void proj_bf16_to_f32(const bf16* __restrict__ A, const float* __restrict__ W,
                      const float* __restrict__ bias, float* __restrict__ out) {
  int lane = threadIdx.x & 31;
  int wave = threadIdx.x >> 5;
  int gw   = blockIdx.x * 8 + wave;
  int tn   = gw & 3;
  int tm   = gw >> 2;

  v8f acc[4] = {{}, {}, {}, {}};
#pragma unroll
  for (int k0 = 0; k0 < C_DIM; k0 += 32) {
    v16bf af = load_a_bf16(A + (size_t)tm * 16 * C_DIM + k0, C_DIM, lane);
#pragma unroll
    for (int s = 0; s < 4; ++s) {
      v16bf bf = load_b_rowmajor_f32(W + (size_t)(tn * 64 + s * 16) * C_DIM + k0,
                                     C_DIM, lane);
      acc[s] = wmma_bf16(af, bf, acc[s]);
    }
  }

  int n  = lane & 15;
  int hi = lane >> 4;
#pragma unroll
  for (int s = 0; s < 4; ++s) {
    int col  = tn * 64 + s * 16 + n;
    float bv = bias ? bias[col] : 0.0f;
#pragma unroll
    for (int r = 0; r < 8; ++r) {
      int m = tm * 16 + r + 8 * hi;
      out[(size_t)m * C_DIM + col] = acc[s][r] + bv;
    }
  }
}

// ---------------------------------------------------------------------------
// Kernel 3: RoPE, in-place on bf16 tensor (B*N, C). One thread per rotation
// pair (reads both halves, writes both) -> in-place is hazard free.
// ---------------------------------------------------------------------------
__global__ __launch_bounds__(256)
void rope_inplace(bf16* __restrict__ t) {
  int idx = blockIdx.x * blockDim.x + threadIdx.x;
  int j   = idx & 15;
  int h   = (idx >> 4) & (NHEADS - 1);
  int row = idx >> 7;
  if (row >= MTOT) return;
  int pos = row % NSEQ;

  float inv_freq = __powf(10000.0f, -(float)j * (1.0f / 16.0f));
  float ang = (float)pos * inv_freq;
  float c, s;
  __sincosf(ang, &c, &s);

  bf16* p = t + (size_t)row * C_DIM + h * HDIM;
  float t1 = (float)p[j];
  float t2 = (float)p[j + 16];
  p[j]      = (bf16)(t1 * c - t2 * s);
  p[j + 16] = (bf16)(t2 * c + t1 * s);
}

// ---------------------------------------------------------------------------
// Kernel 4: flash attention with async double-buffered K/V staging in LDS.
// Block = 8 waves / 128 query rows per (b,h). Keys streamed in tiles of 32;
// each tile fetched ONCE per block via GLOBAL_LOAD_ASYNC_TO_LDS_B128 and
// shared by all 8 waves (8x global-traffic reduction vs per-wave loads).
// ---------------------------------------------------------------------------
__global__ __launch_bounds__(256)
void flash_attn(const bf16* __restrict__ qb, const bf16* __restrict__ kb,
                const bf16* __restrict__ vb, bf16* __restrict__ ob) {
  __shared__ bf16 kTile[2][32 * 32];   // [buf][key][d]
  __shared__ bf16 vTile[2][32 * 32];   // [buf][key][d]
  __shared__ bf16 pTile[8][16 * 32];   // per-wave P re-layout scratch

  int tid  = threadIdx.x;
  int lane = tid & 31;
  int wave = tid >> 5;
  int qblk = blockIdx.x % (NSEQ / 128);
  int bh   = blockIdx.x / (NSEQ / 128);
  int b    = bh / NHEADS;
  int h    = bh % NHEADS;
  int q0   = qblk * 128 + wave * 16;

  const bf16* kbase = kb + (size_t)b * NSEQ * C_DIM + h * HDIM;
  const bf16* vbase = vb + (size_t)b * NSEQ * C_DIM + h * HDIM;

  // Copy roles: threads 0-127 fetch the K tile, 128-255 the V tile.
  // Each thread moves one 16B chunk (128 thr x 16B = 2KB = 32x32 bf16).
  int  crow = (tid & 127) >> 2;
  int  ccol = (tid & 3) * 8;
  bool doK  = tid < 128;

  auto issue_copy = [&](int key, int buf) {
    const bf16* g = (doK ? kbase : vbase) + (size_t)(key + crow) * C_DIM + ccol;
    bf16* l = (doK ? kTile[buf] : vTile[buf]) + crow * 32 + ccol;
    uint32_t loff = (uint32_t)(uintptr_t)l;   // LDS aperture uses addr[31:0]
    asm volatile("global_load_async_to_lds_b128 %0, %1, off"
                 :: "v"(loff), "v"(g) : "memory");
  };

  issue_copy(0, 0);   // prefetch first tile

  const bf16* qptr = qb + (size_t)(b * NSEQ + q0) * C_DIM + h * HDIM;
  v16bf qa = load_a_bf16(qptr, C_DIM, lane);

  v8f o0 = {}, o1 = {};
  float mrow[8], lrow[8];
#pragma unroll
  for (int r = 0; r < 8; ++r) { mrow[r] = -1e30f; lrow[r] = 0.0f; }

  const float scale = 0.17677669529663688f;   // 1/sqrt(32)
  bf16* myp = pTile[wave];
  int n  = lane & 15;
  int hi = lane >> 4;

  for (int kt = 0; kt < NSEQ / 32; ++kt) {
    int buf = kt & 1;
    asm volatile("s_wait_asynccnt 0x0" ::: "memory");
    __syncthreads();                                  // tile `kt` visible; prev reads done
    if (kt + 1 < NSEQ / 32) issue_copy((kt + 1) * 32, buf ^ 1);

    v16bf kb0 = load_b_rowmajor_bf16(kTile[buf], 32, lane);            // keys 0..15
    v16bf kb1 = load_b_rowmajor_bf16(kTile[buf] + 16 * 32, 32, lane);  // keys 16..31
    v8f z = {};
    v8f s0 = wmma_bf16(qa, kb0, z);
    v8f s1 = wmma_bf16(qa, kb1, z);

#pragma unroll
    for (int r = 0; r < 8; ++r) {
      float a0 = s0[r] * scale;
      float a1 = s1[r] * scale;
      float vmax = fmaxf(a0, a1);
#pragma unroll
      for (int off = 1; off < 16; off <<= 1)
        vmax = fmaxf(vmax, __shfl_xor(vmax, off, 32));
      float mnew  = fmaxf(mrow[r], vmax);
      float alpha = __expf(mrow[r] - mnew);
      mrow[r] = mnew;
      float p0 = __expf(a0 - mnew);
      float p1 = __expf(a1 - mnew);
      float ps = p0 + p1;
#pragma unroll
      for (int off = 1; off < 16; off <<= 1)
        ps += __shfl_xor(ps, off, 32);
      lrow[r] = lrow[r] * alpha + ps;
      o0[r] *= alpha;
      o1[r] *= alpha;
      int m = r + 8 * hi;
      myp[m * 32 + n]      = (bf16)p0;      // C-frag -> LDS
      myp[m * 32 + 16 + n] = (bf16)p1;
    }
    asm volatile("s_wait_dscnt 0x0" ::: "memory");    // same-wave P visibility

    v16bf pa  = load_a_bf16(myp, 32, lane);                       // P as A-frag
    v16bf vb0 = load_b_colmajor_bf16(vTile[buf], 32, lane);       // d 0..15
    v16bf vb1 = load_b_colmajor_bf16(vTile[buf] + 16, 32, lane);  // d 16..31
    o0 = wmma_bf16(pa, vb0, o0);
    o1 = wmma_bf16(pa, vb1, o1);
  }

  bf16* outp = ob + (size_t)(b * NSEQ + q0) * C_DIM + h * HDIM;
#pragma unroll
  for (int r = 0; r < 8; ++r) {
    int m = r + 8 * hi;
    float inv = 1.0f / lrow[r];
    outp[(size_t)m * C_DIM + n]      = (bf16)(o0[r] * inv);
    outp[(size_t)m * C_DIM + 16 + n] = (bf16)(o1[r] * inv);
  }
}

// ---------------------------------------------------------------------------
// Host launcher
// ---------------------------------------------------------------------------
extern "C" void kernel_launch(void* const* d_in, const int* in_sizes, int n_in,
                              void* d_out, int out_size, void* d_ws, size_t ws_size,
                              hipStream_t stream) {
  const float* x      = (const float*)d_in[0];
  const float* Wq     = (const float*)d_in[1];
  const float* q_bias = (const float*)d_in[2];
  const float* Wk     = (const float*)d_in[3];
  const float* Wv     = (const float*)d_in[4];
  const float* v_bias = (const float*)d_in[5];
  const float* Wo     = (const float*)d_in[6];
  const float* bo     = (const float*)d_in[7];
  float* out = (float*)d_out;

  const size_t nElem = (size_t)MTOT * C_DIM;   // 2,359,296
  bf16* qbf = (bf16*)d_ws;
  bf16* kbf = qbf + nElem;
  bf16* vbf = kbf + nElem;
  bf16* obf = vbf + nElem;                     // ~18.9 MB of ws

  const int projBlocks  = (MTOT / 16) * (C_DIM / 64) / 8;   // 288
  const int ropeBlocks  = (MTOT * NHEADS * 16) / 256;       // 4608
  const int flashBlocks = BATCH * NHEADS * (NSEQ / 128);    // 576

  proj_f32_to_bf16<<<projBlocks, 256, 0, stream>>>(x, Wq, q_bias, qbf);
  proj_f32_to_bf16<<<projBlocks, 256, 0, stream>>>(x, Wk, nullptr, kbf);
  proj_f32_to_bf16<<<projBlocks, 256, 0, stream>>>(x, Wv, v_bias, vbf);

  rope_inplace<<<ropeBlocks, 256, 0, stream>>>(qbf);
  rope_inplace<<<ropeBlocks, 256, 0, stream>>>(kbf);

  flash_attn<<<flashBlocks, 256, 0, stream>>>(qbf, kbf, vbf, obf);

  proj_bf16_to_f32<<<projBlocks, 256, 0, stream>>>(obf, Wo, bo, out);
}
